// CausalSelfAttention_81277961110148
// MI455X (gfx1250) — matrix operand order repaired and checked by reading, not compiled, not run on hardware
//
#include <hip/hip_runtime.h>
#include <hip/hip_bf16.h>

// ---------------------------------------------------------------------------
// CDNA5 (gfx1250) causal self-attention, f16 WMMA path with f32 accumulation.
//   B=4, T=2048, C=2048, H=16, HD=128
// GEMMs: 256x64 block tiles, 8 waves of 32x64, double-buffered LDS staged via
// GLOBAL_LOAD_ASYNC_TO_LDS_B128 (ASYNCcnt) when the builtin is available.
// ---------------------------------------------------------------------------

typedef __attribute__((ext_vector_type(16))) _Float16 v16h;
typedef __attribute__((ext_vector_type(8)))  _Float16 v8h;
typedef __attribute__((ext_vector_type(8)))  float    v8f;

#define Bb 4
#define Tt 2048
#define Cc 2048
#define Hh 16
#define HD 128

#if __has_builtin(__builtin_amdgcn_global_load_async_to_lds_b128)
#define HAVE_ASYNC 1
typedef int v4i __attribute__((vector_size(16)));   // matches builtin param
typedef __attribute__((address_space(1))) v4i* gas_v4i;
typedef __attribute__((address_space(3))) v4i* las_v4i;
static __device__ __forceinline__ void async_cp16(const _Float16* g,
                                                  _Float16* l) {
  // generic -> as(1): identical 64-bit representation for global memory.
  // generic -> as(3): LDS generic addr = {SHARED_BASE, offset[31:0]}; the
  // low 32 bits are the hardware LDS offset (ISA 10.2 aperture rules).
  __builtin_amdgcn_global_load_async_to_lds_b128(
      (gas_v4i)(unsigned long long)(const void*)g,
      (las_v4i)(unsigned int)(unsigned long long)(void*)l, 0, 0);
}
#define STAGE_CP(g, l) async_cp16((g), (_Float16*)(l))
#else
#define HAVE_ASYNC 0
#define STAGE_CP(g, l) ((void)0)
#endif

#if __has_builtin(__builtin_amdgcn_s_wait_asynccnt)
#define WAIT_ASYNC(n) __builtin_amdgcn_s_wait_asynccnt(n)
#else
#define WAIT_ASYNC(n) asm volatile("s_wait_asynccnt " #n ::: "memory")
#endif

static __device__ __forceinline__ v16h cat8(v8h lo, v8h hi) {
  return __builtin_shufflevector(lo, hi, 0,1,2,3,4,5,6,7,8,9,10,11,12,13,14,15);
}
static __device__ __forceinline__ v8f wmma_f16(v16h a, v16h b, v8f c) {
  // v_wmma_f32_16x16x32_f16  D = A(16x32 f16) * B(32x16 f16) + C(16x16 f32)
  return __builtin_amdgcn_wmma_f32_16x16x32_f16(false, a, false, b, (short)0, c,
                                                false, false);
}

// ---------------------------------------------------------------------------
// fp32 -> f16 conversion (row-major preserved), 8 elements / thread.
// ---------------------------------------------------------------------------
__global__ void cvt_f32_f16(const float* __restrict__ in,
                            _Float16* __restrict__ out, int n8) {
  int i = blockIdx.x * blockDim.x + threadIdx.x;
  if (i >= n8) return;
  const float4* p = (const float4*)in + (size_t)i * 2;
  float4 a = p[0], b = p[1];
  v8h o;
  o[0] = (_Float16)a.x; o[1] = (_Float16)a.y;
  o[2] = (_Float16)a.z; o[3] = (_Float16)a.w;
  o[4] = (_Float16)b.x; o[5] = (_Float16)b.y;
  o[6] = (_Float16)b.z; o[7] = (_Float16)b.w;
  *((v8h*)out + i) = o;
}

// ---------------------------------------------------------------------------
// fp32 [K][N] -> f16 [N][K] transposed conversion via 64x64 LDS tiles,
// coalesced on both sides.  Lets GEMM B-tiles be contiguous async copies.
// ---------------------------------------------------------------------------
__global__ __launch_bounds__(256) void cvt_transpose_f16(
    const float* __restrict__ in, _Float16* __restrict__ out, int K, int N) {
  __shared__ _Float16 t[64][72];
  int k0 = blockIdx.y * 64, n0 = blockIdx.x * 64;
  int tr = threadIdx.x >> 2;        // 0..63
  int tc = (threadIdx.x & 3) * 16;  // 0,16,32,48
  const float* src = in + (size_t)(k0 + tr) * N + n0 + tc;
#pragma unroll
  for (int i = 0; i < 16; ++i) t[tr][tc + i] = (_Float16)src[i];
  __syncthreads();
  _Float16* dst = out + (size_t)(n0 + tr) * K + k0 + tc;
#pragma unroll
  for (int i = 0; i < 16; ++i) dst[i] = t[tc + i][tr];
}

// ---------------------------------------------------------------------------
// Double-buffered staging of one (TM x 32) A tile + (64 x 32) B tile.
//   A: thread 'tid' owns row tid, copies 64B (4 x b128).
//   B: thread owns 16B of Wt (pre-transposed weights, [N][K]).
// ---------------------------------------------------------------------------
#define TM 256
#define TN 64
#define NKS (Cc / 32)  // 64 k-steps

#define STAGE(it)                                                         \
  {                                                                       \
    int sb = (it) & 1;                                                    \
    int k0 = (it) * 32;                                                   \
    if (HAVE_ASYNC) {                                                     \
      _Pragma("unroll") for (int j = 0; j < 4; ++j)                       \
          STAGE_CP(aSrc + k0 + j * 8, &As[sb][tid][j * 8]);               \
      STAGE_CP(bSrc + k0, &Bs[sb][bRow][bCol]);                           \
    } else {                                                              \
      _Pragma("unroll") for (int j = 0; j < 4; ++j)                       \
          *(v8h*)&As[sb][tid][j * 8] = *(const v8h*)(aSrc + k0 + j * 8);  \
      *(v8h*)&Bs[sb][bRow][bCol] = *(const v8h*)(bSrc + k0);              \
    }                                                                     \
  }

// ---------------------------------------------------------------------------
// QKV projection GEMM (M=8192,K=2048,N=6144) + bias + RoPE(k,v) epilogue.
// ---------------------------------------------------------------------------
__global__ __launch_bounds__(256) void qkv_rope_kernel(
    const _Float16* __restrict__ xh,    // [8192][2048]
    const _Float16* __restrict__ Wat,   // [6144][2048]  (transposed)
    const float*    __restrict__ bias,  // [6144]
    const float*    __restrict__ fcos,  // [2048][64]
    const float*    __restrict__ fsin,  // [2048][64]
    _Float16* __restrict__ qh, _Float16* __restrict__ kh,
    _Float16* __restrict__ vh) {
  const int K = Cc;
  __shared__ _Float16 As[2][TM][40];
  __shared__ _Float16 Bs[2][TN][40];

  const int M0 = blockIdx.y * TM;
  const int N0 = blockIdx.x * TN;
  const int tid = threadIdx.x, lane = tid & 31, wid = tid >> 5;
  const int n_l = lane & 15;
  const int kb_a = (lane < 16) ? 0 : 8;   // A-frag K base (ISA layout)
  const int kb_b = (lane < 16) ? 0 : 8;   // B-frag K base
  const int mh = (lane >> 4) << 3;        // C-frag row offset
  const int wm = wid * 32;

  const _Float16* aSrc = xh + (size_t)(M0 + tid) * K;
  const int bRow = tid >> 2, bCol = (tid & 3) * 8;
  const _Float16* bSrc = Wat + (size_t)(N0 + bRow) * K + bCol;

  v8f acc[2][4] = {};

  STAGE(0);
  for (int it = 0; it < NKS; ++it) {
    const int buf = it & 1;
    if (it + 1 < NKS) {
      STAGE(it + 1);
#if HAVE_ASYNC
      WAIT_ASYNC(5);  // previous stage's 5 copies have landed (in-order)
#endif
    } else {
#if HAVE_ASYNC
      WAIT_ASYNC(0);
#endif
    }
    __syncthreads();

    v16h a[2], b[4];
#pragma unroll
    for (int i = 0; i < 2; ++i) {
      const _Float16* p = &As[buf][wm + i * 16 + n_l][0];
      a[i] = cat8(*(const v8h*)(p + kb_a), *(const v8h*)(p + kb_a + 16));
    }
#pragma unroll
    for (int j = 0; j < 4; ++j) {
      const _Float16* p = &Bs[buf][j * 16 + n_l][0];
      b[j] = cat8(*(const v8h*)(p + kb_b), *(const v8h*)(p + kb_b + 16));
    }
#pragma unroll
    for (int i = 0; i < 2; ++i)
#pragma unroll
      for (int j = 0; j < 4; ++j) acc[i][j] = wmma_f16(a[i], b[j], acc[i][j]);
    __syncthreads();
  }

  // Epilogue: bias + RoPE(k,v) + scatter to (B,H,T,HD) f16
#pragma unroll
  for (int i = 0; i < 2; ++i)
#pragma unroll
    for (int j = 0; j < 4; ++j) {
      int cg = N0 + j * 16 + n_l;
      float bv = bias[cg];
      int sec = cg >> 11;  // 0:q 1:k 2:v (uniform per wave)
      int cc = cg & (Cc - 1);
      int h = cc >> 7, d = cc & (HD - 1);
      _Float16* dst = (sec == 0) ? qh : ((sec == 1) ? kh : vh);
#pragma unroll
      for (int r = 0; r < 8; ++r) {
        int mg = M0 + wm + i * 16 + mh + r;  // global row = b*T + t
        int bb = mg >> 11, t = mg & (Tt - 1);
        float v = acc[i][j][r] + bv;
        float partner = __shfl_xor(v, 1, 32);  // RoPE pair partner
        float outv = v;
        if (sec != 0) {
          float cf = fcos[t * (HD / 2) + (d >> 1)];
          float sf = fsin[t * (HD / 2) + (d >> 1)];
          outv = (d & 1) ? (partner * sf + v * cf) : (v * cf - partner * sf);
        }
        dst[(((size_t)bb * Hh + h) * Tt + t) * HD + d] = (_Float16)outv;
      }
    }
}

// ---------------------------------------------------------------------------
// Flash attention, one (b,h) per blockIdx.y, 64 query rows per block,
// 4 waves x 16 rows each, fully wave-local.  Key blocks of 32; K fragments
// loaded straight from global (d-contiguous = B-fragment layout); V transposed
// via per-wave private LDS; online softmax in fp32.
// ---------------------------------------------------------------------------
__global__ __launch_bounds__(128) void attn_kernel(
    const _Float16* __restrict__ qh, const _Float16* __restrict__ kh,
    const _Float16* __restrict__ vh, _Float16* __restrict__ ah) {
  __shared__ _Float16 Vt[4][HD][40];  // per-wave V^T tile [d][key]
  __shared__ _Float16 Sw[4][16][40];  // per-wave P tile   [m][key]

  const int bh = blockIdx.y;
  const int wid = threadIdx.x >> 5, lane = threadIdx.x & 31;
  const int q0 = blockIdx.x * 64 + wid * 16;
  const _Float16* Q  = qh + (size_t)bh * Tt * HD;
  const _Float16* Kp = kh + (size_t)bh * Tt * HD;
  const _Float16* Vp = vh + (size_t)bh * Tt * HD;

  const int n_l = lane & 15;
  const int mh = (lane >> 4) << 3;
  const int kb_a = (lane < 16) ? 0 : 8;
  const int kb_b = (lane < 16) ? 0 : 8;
  const float scale = 0.08838834764831845f;  // 1/sqrt(128)

  v16h qa[4];
  {
    const _Float16* qrow = Q + (size_t)(q0 + n_l) * HD;
#pragma unroll
    for (int f = 0; f < 4; ++f)
      qa[f] = cat8(*(const v8h*)(qrow + f * 32 + kb_a),
                   *(const v8h*)(qrow + f * 32 + kb_a + 16));
  }

  v8f o[8] = {};
  float rmax[8], rsum[8];
#pragma unroll
  for (int r = 0; r < 8; ++r) { rmax[r] = -__builtin_inff(); rsum[r] = 0.f; }

  for (int kb = 0; kb <= q0 + 15; kb += 32) {
    {  // stage V^T: lane owns key 'lane' of this block
      const _Float16* vrow = Vp + (size_t)(kb + lane) * HD;
#pragma unroll
      for (int c = 0; c < HD; c += 8) {
        v8h vv = *(const v8h*)(vrow + c);
#pragma unroll
        for (int i = 0; i < 8; ++i) Vt[wid][c + i][lane] = vv[i];
      }
    }

    // S = Q K^T : two 16x16 f32 tiles (keys kb..kb+15, kb+16..kb+31)
    v8f s0 = {}, s1 = {};
#pragma unroll
    for (int f = 0; f < 4; ++f) {
      const _Float16* k0p = Kp + (size_t)(kb + n_l) * HD + f * 32 + kb_b;
      const _Float16* k1p = k0p + (size_t)16 * HD;
      v16h bk0 = cat8(*(const v8h*)k0p, *(const v8h*)(k0p + 16));
      v16h bk1 = cat8(*(const v8h*)k1p, *(const v8h*)(k1p + 16));
      s0 = wmma_f16(qa[f], bk0, s0);
      s1 = wmma_f16(qa[f], bk1, s1);
    }

    // scale + causal mask + online softmax update
    float corr[8];
#pragma unroll
    for (int r = 0; r < 8; ++r) {
      int mrow = q0 + mh + r;
      float a0 = s0[r] * scale;
      float a1 = s1[r] * scale;
      if (kb + n_l > mrow)      a0 = -__builtin_inff();
      if (kb + 16 + n_l > mrow) a1 = -__builtin_inff();
      float ml = fmaxf(a0, a1);
#pragma unroll
      for (int off = 1; off <= 8; off <<= 1)
        ml = fmaxf(ml, __shfl_xor(ml, off, 32));
      float nm = fmaxf(rmax[r], ml);
      float c = __expf(rmax[r] - nm);
      rmax[r] = nm;
      float p0 = __expf(a0 - nm);
      float p1 = __expf(a1 - nm);
      float ls = p0 + p1;
#pragma unroll
      for (int off = 1; off <= 8; off <<= 1) ls += __shfl_xor(ls, off, 32);
      rsum[r] = rsum[r] * c + ls;
      corr[r] = c;
      Sw[wid][mh + r][n_l]      = (_Float16)p0;
      Sw[wid][mh + r][16 + n_l] = (_Float16)p1;
    }
#pragma unroll
    for (int f = 0; f < 8; ++f)
#pragma unroll
      for (int r = 0; r < 8; ++r) o[f][r] *= corr[r];

    // same-wave LDS write -> cross-lane-read ordering
    asm volatile("s_wait_dscnt 0" ::: "memory");

    // P fragment (16x32) from Sw, then O += P * V
    const _Float16* prow = &Sw[wid][n_l][0];
    v16h pa = cat8(*(const v8h*)(prow + kb_a), *(const v8h*)(prow + kb_a + 16));
#pragma unroll
    for (int f = 0; f < 8; ++f) {
      const _Float16* vtp = &Vt[wid][f * 16 + n_l][kb_b];
      v16h vb = cat8(*(const v8h*)vtp, *(const v8h*)(vtp + 16));
      o[f] = wmma_f16(pa, vb, o[f]);
    }
  }

  // normalize + store attn output as (B,T,C) f16
  const int b = bh >> 4, h = bh & 15;
#pragma unroll
  for (int r = 0; r < 8; ++r) {
    float inv = 1.0f / rsum[r];
    int t = q0 + mh + r;
    _Float16* dst = ah + ((size_t)b * Tt + t) * Cc + h * HD;
#pragma unroll
    for (int f = 0; f < 8; ++f)
      dst[f * 16 + n_l] = (_Float16)(o[f][r] * inv);
  }
}

// ---------------------------------------------------------------------------
// Output projection GEMM (M=8192,K=2048,N=2048) + bias, fp32 output.
// ---------------------------------------------------------------------------
__global__ __launch_bounds__(256) void proj_kernel(
    const _Float16* __restrict__ ah,   // [8192][2048] f16
    const _Float16* __restrict__ Wpt,  // [2048][2048] f16 (transposed)
    const float* __restrict__ bias,    // [2048]
    float* __restrict__ out) {         // [8192][2048] fp32
  const int K = Cc;
  __shared__ _Float16 As[2][TM][40];
  __shared__ _Float16 Bs[2][TN][40];

  const int M0 = blockIdx.y * TM;
  const int N0 = blockIdx.x * TN;
  const int tid = threadIdx.x, lane = tid & 31, wid = tid >> 5;
  const int n_l = lane & 15;
  const int kb_a = (lane < 16) ? 0 : 8;
  const int kb_b = (lane < 16) ? 0 : 8;
  const int mh = (lane >> 4) << 3;
  const int wm = wid * 32;

  const _Float16* aSrc = ah + (size_t)(M0 + tid) * K;
  const int bRow = tid >> 2, bCol = (tid & 3) * 8;
  const _Float16* bSrc = Wpt + (size_t)(N0 + bRow) * K + bCol;

  v8f acc[2][4] = {};

  STAGE(0);
  for (int it = 0; it < NKS; ++it) {
    const int buf = it & 1;
    if (it + 1 < NKS) {
      STAGE(it + 1);
#if HAVE_ASYNC
      WAIT_ASYNC(5);
#endif
    } else {
#if HAVE_ASYNC
      WAIT_ASYNC(0);
#endif
    }
    __syncthreads();

    v16h a[2], b[4];
#pragma unroll
    for (int i = 0; i < 2; ++i) {
      const _Float16* p = &As[buf][wm + i * 16 + n_l][0];
      a[i] = cat8(*(const v8h*)(p + kb_a), *(const v8h*)(p + kb_a + 16));
    }
#pragma unroll
    for (int j = 0; j < 4; ++j) {
      const _Float16* p = &Bs[buf][j * 16 + n_l][0];
      b[j] = cat8(*(const v8h*)(p + kb_b), *(const v8h*)(p + kb_b + 16));
    }
#pragma unroll
    for (int i = 0; i < 2; ++i)
#pragma unroll
      for (int j = 0; j < 4; ++j) acc[i][j] = wmma_f16(a[i], b[j], acc[i][j]);
    __syncthreads();
  }

#pragma unroll
  for (int i = 0; i < 2; ++i)
#pragma unroll
    for (int j = 0; j < 4; ++j) {
      int cg = N0 + j * 16 + n_l;
      float bv = bias[cg];
#pragma unroll
      for (int r = 0; r < 8; ++r) {
        int mg = M0 + wm + i * 16 + mh + r;
        out[(size_t)mg * Cc + cg] = acc[i][j][r] + bv;
      }
    }
}

// ---------------------------------------------------------------------------
extern "C" void kernel_launch(void* const* d_in, const int* in_sizes, int n_in,
                              void* d_out, int out_size, void* d_ws,
                              size_t ws_size, hipStream_t stream) {
  const float* x      = (const float*)d_in[0];
  const float* fcos   = (const float*)d_in[1];
  const float* fsin   = (const float*)d_in[2];
  // d_in[3] = mask (causality computed analytically, unused)
  const float* W_attn = (const float*)d_in[4];
  const float* b_attn = (const float*)d_in[5];
  const float* W_proj = (const float*)d_in[6];
  const float* b_proj = (const float*)d_in[7];
  float* out = (float*)d_out;

  // workspace layout (f16), total ~192 MB
  _Float16* xh  = (_Float16*)d_ws;                  // 8192*2048
  _Float16* Wat = xh  + (size_t)8192 * 2048;        // 6144*2048 (transposed)
  _Float16* Wpt = Wat + (size_t)6144 * 2048;        // 2048*2048 (transposed)
  _Float16* qh  = Wpt + (size_t)2048 * 2048;        // B*H*T*HD
  _Float16* kh  = qh  + (size_t)Bb * Hh * Tt * HD;
  _Float16* vh  = kh  + (size_t)Bb * Hh * Tt * HD;
  _Float16* ah  = vh  + (size_t)Bb * Hh * Tt * HD;  // B*T*C

  {
    int n8 = (int)((size_t)8192 * 2048 / 8);
    cvt_f32_f16<<<(n8 + 255) / 256, 256, 0, stream>>>(x, xh, n8);
  }
  cvt_transpose_f16<<<dim3(6144 / 64, 2048 / 64), 256, 0, stream>>>(
      W_attn, Wat, 2048, 6144);
  cvt_transpose_f16<<<dim3(2048 / 64, 2048 / 64), 256, 0, stream>>>(
      W_proj, Wpt, 2048, 2048);

  qkv_rope_kernel<<<dim3(6144 / TN, 8192 / TM), 256, 0, stream>>>(
      xh, Wat, b_attn, fcos, fsin, qh, kh, vh);

  attn_kernel<<<dim3(Tt / 64, Bb * Hh), 128, 0, stream>>>(qh, kh, vh, ah);

  proj_kernel<<<dim3(2048 / TN, 8192 / TM), 256, 0, stream>>>(ah, Wpt, b_proj,
                                                              out);
}
